// HungarianLoss_85323820303020
// MI455X (gfx1250) — compile-verified, matrix operand order checked by reading, not run on hardware
//
#include <hip/hip_runtime.h>
#include <hip/hip_bf16.h>

typedef __attribute__((ext_vector_type(16))) _Float16 v16h;
typedef __attribute__((ext_vector_type(8)))  float    v8f;

#define Bb 32
#define Mm 384
#define Nn 128
#define Cc 92
#define CP 96          // C padded to multiple of 32 for WMMA K
#define NTHREADS 384   // 12 wave32s; one thread per pred column
#define NWAVES (NTHREADS / 32)
#define W_IOU 2.0f
#define W_L1  5.0f
#define W_CAT 1.0f
#define EPSf 1e-7f
#define BIG 1e30f

// ---- LDS layout (bytes). Total 315712 <= 320KB/WGP (CDNA5-only capacity). ----
#define OFF_COST 0        // 128*384*4 = 196608 : whole cost matrix stays on-chip
#define OFF_E    196608   // 384*96*2  = 73728  : exp(logit - rowmax) in f16 (WMMA B)
#define OFF_OH   270336   // 128*96*2  = 24576  : one-hot(targ_cat) in f16 (WMMA A)
#define OFF_PXY  294912   // 384*16             : pred boxes xyxy
#define OFF_TXY  301056   // 128*16             : targ boxes xyxy
#define OFF_RCZ  303104   // 384*4              : 1/Z per pred row
#define OFF_LZM  304640   // 384*4              : max + log(Z) per pred row
#define OFF_V    306176   // 384*4              : column potentials
#define OFF_MINV 307712   // 384*4
#define OFF_WAY  309248   // 384*4
#define OFF_USED 310784   // 384*4
#define OFF_P    312320   // 384*4              : row assigned to column (-1 free)
#define OFF_U    313856   // 128*4              : row potentials
#define OFF_TC   314368   // 128*4
#define OFF_ASG  314880   // 128*4
#define OFF_R0   315392   // 16*4 reduction partials
#define OFF_R1   315456
#define OFF_R2   315520
#define OFF_RI   315584
#define OFF_SC   315648   // scalars: int j0,i0,j1,done ; float delta
#define SMEM_BYTES 315712

__device__ __forceinline__ float pair_cost(float4 p, float4 t) {
    // W_IOU * giou_loss + W_L1 * mean|p-t|
    float ix1 = fmaxf(p.x, t.x), iy1 = fmaxf(p.y, t.y);
    float ix2 = fminf(p.z, t.z), iy2 = fminf(p.w, t.w);
    float inter = fmaxf(ix2 - ix1, 0.f) * fmaxf(iy2 - iy1, 0.f);
    float ap = (p.z - p.x) * (p.w - p.y);
    float at = (t.z - t.x) * (t.w - t.y);
    float uni = ap + at - inter;
    float iou = inter / (uni + EPSf);
    float cx1 = fminf(p.x, t.x), cy1 = fminf(p.y, t.y);
    float cx2 = fmaxf(p.z, t.z), cy2 = fmaxf(p.w, t.w);
    float ac = (cx2 - cx1) * (cy2 - cy1);
    float giou = iou - (ac - uni) / (ac + EPSf);
    float l1 = (fabsf(p.x - t.x) + fabsf(p.y - t.y) +
                fabsf(p.z - t.z) + fabsf(p.w - t.w)) * 0.25f;
    return W_IOU * (1.f - giou) + W_L1 * l1;
}

__global__ void __launch_bounds__(NTHREADS)
hungarian_fused(const float* __restrict__ pred_cat,  // (B,M,C)
                const float* __restrict__ pred_bbox, // (B,M,4) cxcywh
                const int*   __restrict__ targ_cat,  // (B,N)
                const float* __restrict__ targ_bbox, // (B,N,4) cxcywh
                float* __restrict__ out,             // [0]=loss (by finalize), [1..B*N]=assign
                float* __restrict__ batch_loss)      // (B,) in workspace
{
    extern __shared__ char smem[];
    float*    costS = (float*)(smem + OFF_COST);
    _Float16* ES    = (_Float16*)(smem + OFF_E);
    _Float16* ohS   = (_Float16*)(smem + OFF_OH);
    float4*   pxyS  = (float4*)(smem + OFF_PXY);
    float4*   txyS  = (float4*)(smem + OFF_TXY);
    float*    rczS  = (float*)(smem + OFF_RCZ);
    float*    lzmS  = (float*)(smem + OFF_LZM);
    float*    vS    = (float*)(smem + OFF_V);
    float*    minvS = (float*)(smem + OFF_MINV);
    int*      wayS  = (int*)(smem + OFF_WAY);
    int*      usedS = (int*)(smem + OFF_USED);
    int*      pS    = (int*)(smem + OFF_P);
    float*    uS    = (float*)(smem + OFF_U);
    int*      tcS   = (int*)(smem + OFF_TC);
    int*      asgS  = (int*)(smem + OFF_ASG);
    float*    r0    = (float*)(smem + OFF_R0);
    float*    r1    = (float*)(smem + OFF_R1);
    float*    r2    = (float*)(smem + OFF_R2);
    int*      ri    = (int*)(smem + OFF_RI);
    int*      scI   = (int*)(smem + OFF_SC);
    float*    scF   = (float*)(smem + OFF_SC + 16);

    const int b    = blockIdx.x;
    const int tid  = threadIdx.x;
    const int wv   = tid >> 5;
    const int lane = tid & 31;
    const int lr   = lane & 15;        // tile row/col for this lane
    const int hb   = (lane >> 4) * 8;  // K-half base per WMMA f16 layout

    // ---------- Phase 1: per-pred softmax stats + exp table (f16) + xyxy ----------
    {
        const int m = tid; // M == NTHREADS
        const float* lrow = pred_cat + ((size_t)b * Mm + m) * Cc;
        float mx = -BIG;
        for (int c = 0; c < Cc; ++c) mx = fmaxf(mx, lrow[c]);
        float s = 0.f;
        _Float16* erow = ES + m * CP;
        for (int c = 0; c < Cc; ++c) {
            float e = __expf(lrow[c] - mx);
            s += e;
            erow[c] = (_Float16)e;
        }
        for (int c = Cc; c < CP; ++c) erow[c] = (_Float16)0.f;
        rczS[m] = 1.f / s;
        lzmS[m] = mx + __logf(s);
        float4 bb = ((const float4*)pred_bbox)[(size_t)b * Mm + m];
        float4 xy;
        xy.x = bb.x - 0.5f * bb.z; xy.y = bb.y - 0.5f * bb.w;
        xy.z = bb.x + 0.5f * bb.z; xy.w = bb.y + 0.5f * bb.w;
        pxyS[m] = xy;
    }
    // ---------- Phase 2: targets: class, one-hot (f16), xyxy ----------
    if (tid < Nn) {
        const int n = tid;
        const int tc = targ_cat[b * Nn + n];
        tcS[n] = tc;
        float4 tb = ((const float4*)targ_bbox)[(size_t)b * Nn + n];
        float4 xy;
        xy.x = tb.x - 0.5f * tb.z; xy.y = tb.y - 0.5f * tb.w;
        xy.z = tb.x + 0.5f * tb.z; xy.w = tb.y + 0.5f * tb.w;
        txyS[n] = xy;
    }
    for (int k = tid; k < Nn * CP / 2; k += NTHREADS) ((unsigned int*)ohS)[k] = 0u;
    __syncthreads();
    if (tid < Nn) ohS[tid * CP + tcS[tid]] = (_Float16)1.f;
    __syncthreads();

    // ---------- Phase 3: WMMA gather-GEMM + fused cost matrix into LDS ----------
    // G[n,m] = sum_c onehot[n,c] * exp(logit[m,c]-max_m); cost = W_CAT*(1 - G*rz)
    //          + mask(n) * pair_cost. 8x24 tiles of 16x16, K = 96 in 3 steps.
    for (int t = wv; t < (Nn / 16) * (Mm / 16); t += NWAVES) {
        const int ti = t / (Mm / 16);
        const int tj = t % (Mm / 16);
        v8f acc = {0.f, 0.f, 0.f, 0.f, 0.f, 0.f, 0.f, 0.f};
        const _Float16* arow = ohS + (ti * 16 + lr) * CP; // A: one-hot rows (N x K)
        const _Float16* brow = ES  + (tj * 16 + lr) * CP; // B: E^T cols == E rows (K x M-tile)
        #pragma unroll
        for (int kk = 0; kk < CP; kk += 32) {
            v16h af, bf;
            // per-lane f16 A/B layout: halves 0..7 at K=kb.., halves 8..15 at K=kb+16..
            *((uint4*)&af)       = *((const uint4*)(arow + kk + hb));
            *(((uint4*)&af) + 1) = *((const uint4*)(arow + kk + hb + 16));
            *((uint4*)&bf)       = *((const uint4*)(brow + kk + hb));
            *(((uint4*)&bf) + 1) = *((const uint4*)(brow + kk + hb + 16));
            acc = __builtin_amdgcn_wmma_f32_16x16x32_f16(
                false, af, false, bf, (short)0, acc, false, false);
        }
        const int m = tj * 16 + lr;
        const float rz = rczS[m];
        const float4 pbx = pxyS[m];
        const int nb = ti * 16 + hb; // D layout: row n = 8*(lane>=16) + vgpr
        #pragma unroll
        for (int vv = 0; vv < 8; ++vv) {
            const int n = nb + vv;
            float c = W_CAT * (1.f - acc[vv] * rz);
            float msk = (tcS[n] != 0) ? 1.f : 0.f;
            c += msk * pair_cost(pbx, txyS[n]);
            costS[n * Mm + m] = c;
        }
    }
    __syncthreads();

    // ---------- Phase 4: Hungarian (JV shortest augmenting path), cost in LDS ----------
    vS[tid] = 0.f;
    pS[tid] = -1;
    if (tid < Nn) uS[tid] = 0.f;
    __syncthreads();

    for (int i = 0; i < Nn; ++i) {
        minvS[tid] = BIG;
        usedS[tid] = 0;
        wayS[tid]  = -1;
        if (tid == 0) { scI[0] = -1; scI[1] = i; scI[3] = 0; }
        __syncthreads();
        while (true) {
            const int   j0 = scI[0];
            const int   i0 = scI[1];
            const float u0 = uS[i0];
            // relax this thread's column
            float mv = minvS[tid];
            if (!usedS[tid]) {
                float cur = costS[i0 * Mm + tid] - u0 - vS[tid];
                if (cur < mv) { mv = cur; minvS[tid] = cur; wayS[tid] = j0; }
            }
            // argmin over free columns (first-min tie-break like np.argmin)
            float val = usedS[tid] ? BIG : mv;
            int   idx = tid;
            #pragma unroll
            for (int off = 16; off > 0; off >>= 1) {
                float ov = __shfl_xor(val, off, 32);
                int   oi = __shfl_xor(idx, off, 32);
                if (ov < val || (ov == val && oi < idx)) { val = ov; idx = oi; }
            }
            if (lane == 0) { r0[wv] = val; ri[wv] = idx; }
            __syncthreads();
            if (tid == 0) {
                float bv = r0[0]; int bi = ri[0];
                for (int w2 = 1; w2 < NWAVES; ++w2)
                    if (r0[w2] < bv || (r0[w2] == bv && ri[w2] < bi)) { bv = r0[w2]; bi = ri[w2]; }
                scF[0] = bv;  // delta
                scI[2] = bi;  // j1
            }
            __syncthreads();
            const float delta = scF[0];
            const int   j1    = scI[2];
            // dual updates with the *old* used set (j1 still free here)
            if (usedS[tid]) {
                vS[tid] -= delta;
                uS[pS[tid]] += delta;      // distinct rows -> race-free
            } else {
                minvS[tid] -= delta;
            }
            if (tid == j1) usedS[tid] = 1; // becomes used for next iteration
            if (tid == 0) {
                uS[i] += delta;            // sentinel row i on the path
                scI[0] = j1;
                const int pj = pS[j1];
                scI[1] = pj;
                scI[3] = (pj < 0) ? 1 : 0;
            }
            __syncthreads();
            if (scI[3]) break;
        }
        if (tid == 0) { // augment along way[] chain
            int j0 = scI[0];
            while (j0 >= 0) {
                const int jp = wayS[j0];
                pS[j0] = (jp < 0) ? i : pS[jp];
                j0 = jp;
            }
        }
        __syncthreads();
    }

    // ---------- Phase 5: invert assignment, per-batch loss ----------
    if (pS[tid] >= 0) asgS[pS[tid]] = tid;
    __syncthreads();
    float ce = 0.f, bl = 0.f, mk = 0.f;
    if (tid < Nn) {
        const int n = tid;
        const int a = asgS[n];
        const int tc = tcS[n];
        ce = lzmS[a] - pred_cat[((size_t)b * Mm + a) * Cc + tc]; // -log_softmax
        if (tc != 0) { bl = pair_cost(pxyS[a], txyS[n]); mk = 1.f; }
    }
    #pragma unroll
    for (int off = 16; off > 0; off >>= 1) {
        ce += __shfl_xor(ce, off, 32);
        bl += __shfl_xor(bl, off, 32);
        mk += __shfl_xor(mk, off, 32);
    }
    if (lane == 0) { r0[wv] = ce; r1[wv] = bl; r2[wv] = mk; }
    __syncthreads();
    if (tid == 0) {
        float ces = 0.f, bls = 0.f, mks = 0.f;
        for (int w2 = 0; w2 < NWAVES; ++w2) { ces += r0[w2]; bls += r1[w2]; mks += r2[w2]; }
        batch_loss[b] = (ces / (float)Nn) * 1.0f /*LOSS_CAT*/ + bls / (mks + EPSf);
    }
    if (tid < Nn) out[1 + b * Nn + tid] = (float)asgS[tid];
}

// Deterministic fixed-order batch mean (no float atomics).
__global__ void finalize_loss(const float* __restrict__ batch_loss, float* __restrict__ out) {
    if (threadIdx.x == 0) {
        float s = 0.f;
        for (int b = 0; b < Bb; ++b) s += batch_loss[b];
        out[0] = s / (float)Bb;
    }
}

extern "C" void kernel_launch(void* const* d_in, const int* in_sizes, int n_in,
                              void* d_out, int out_size, void* d_ws, size_t ws_size,
                              hipStream_t stream) {
    (void)in_sizes; (void)n_in; (void)out_size; (void)ws_size;
    const float* pred_cat  = (const float*)d_in[0];
    const float* pred_bbox = (const float*)d_in[1];
    const int*   targ_cat  = (const int*)d_in[2];
    const float* targ_bbox = (const float*)d_in[3];
    float* out = (float*)d_out;
    float* batch_loss = (float*)d_ws; // 32 floats of scratch

    // Opt in to >64KB dynamic LDS (CDNA5 WGP has 320KB).
    (void)hipFuncSetAttribute(reinterpret_cast<const void*>(hungarian_fused),
                              hipFuncAttributeMaxDynamicSharedMemorySize, SMEM_BYTES);

    hungarian_fused<<<Bb, NTHREADS, SMEM_BYTES, stream>>>(
        pred_cat, pred_bbox, targ_cat, targ_bbox, out, batch_loss);
    finalize_loss<<<1, 32, 0, stream>>>(batch_loss, out);
}
